// SparseAttention_28922309771642
// MI455X (gfx1250) — matrix operand order repaired and checked by reading, not compiled, not run on hardware
//
#include <hip/hip_runtime.h>

typedef __attribute__((ext_vector_type(16))) _Float16 v16h;
typedef __attribute__((ext_vector_type(8)))  float    v8f;

#define B_ 8
#define S_ 4096
#define E_ 16
#define D_ 128
#define G_ 256
#define SCALE 0.08838834764831845f   // 1/sqrt(128)

// ---------------------------------------------------------------------------
// Phase 1a: expert assignment = argmax of one-hot route matrix.
// ---------------------------------------------------------------------------
__global__ void assign_kernel(const float* __restrict__ route, int* __restrict__ assign) {
    int g = blockIdx.x * blockDim.x + threadIdx.x;     // 0 .. B*S-1
    if (g >= B_ * S_) return;
    const float* r = route + (size_t)g * E_;
    float best = -1.0f; int bi = 0;
    #pragma unroll
    for (int e = 0; e < E_; ++e) {
        float v = r[e];
        if (v > best) { best = v; bi = e; }
    }
    assign[g] = bi;
}

// ---------------------------------------------------------------------------
// Phase 1b: stable per-(b,e) compaction of token indices (== stable argsort
// by expert id). One wave32 per (b,e); ballot + prefix popcount is stable.
// ---------------------------------------------------------------------------
__global__ void build_idx_kernel(const int* __restrict__ assign, int* __restrict__ idx) {
    int bid  = blockIdx.x;                 // b*E + e
    int b    = bid >> 4;
    int e    = bid & (E_ - 1);
    int lane = threadIdx.x & 31;
    const int* ab = assign + b * S_;
    int* out = idx + bid * G_;
    int count = 0;
    for (int base = 0; base < S_; base += 32) {
        int s = base + lane;
        bool pred = (ab[s] == e);
        unsigned mask = (unsigned)__ballot(pred);      // wave32: low 32 bits
        if (pred) {
            int rank = __popc(mask & ((1u << lane) - 1u));
            out[count + rank] = s;
        }
        count += __popc(mask);
    }
}

// ---------------------------------------------------------------------------
// WMMA helpers (layouts per CDNA5 ISA §7.12.2)
// ---------------------------------------------------------------------------
__device__ __forceinline__ v8f wmma16(v16h a, v16h b, v8f c) {
    return __builtin_amdgcn_wmma_f32_16x16x32_f16(
        /*neg_a=*/false, a, /*neg_b=*/false, b,
        /*c_mod=*/(short)0, c, /*reuse_a=*/false, /*reuse_b=*/false);
}

// 16-bit A 16x32 in this code's contiguous k order (B matches): lanes<16 hold K = kc+0..15 for row M=ln;
// lanes>=16 hold K = kc+16..31.
__device__ __forceinline__ v16h load_a16x32(const _Float16* base, int row, int stride,
                                            int kc, int halfsel) {
    const _Float16* p = base + row * stride + kc + halfsel * 16;
    v16h a;
    #pragma unroll
    for (int i = 0; i < 16; ++i) a[i] = p[i];
    return a;
}

// B fragment for S = Q*K^T : B[k,n] = Ke[nbase+n, c*32+k], contiguous in k.
__device__ __forceinline__ v16h load_bK(const _Float16* Kf, int nbase, int n,
                                        int c, int halfsel) {
    const _Float16* p = Kf + (nbase + n) * D_ + c * 32 + halfsel * 16;
    v16h bf;
    #pragma unroll
    for (int i = 0; i < 16; ++i) bf[i] = p[i];
    return bf;
}

// B fragment for X = P*V : B[k,n] = Ve[k0+k, dcol+n], stride D_ in k.
__device__ __forceinline__ v16h load_bV(const _Float16* Vf, int k0, int dcol,
                                        int n, int halfsel) {
    const _Float16* p = Vf + (k0 + halfsel * 16) * D_ + dcol + n;
    v16h bf;
    #pragma unroll
    for (int i = 0; i < 16; ++i) bf[i] = p[i * D_];
    return bf;
}

// ---------------------------------------------------------------------------
// Phase 2: per-(b,e) flash attention out of LDS. 256 threads = 8 waves,
// each wave owns query tiles {wave, wave+8}. Dynamic LDS: Kf/Vf/Qf (64KB each,
// f16) + 1KB/wave P staging = 200KB < 320KB WGP LDS.
// ---------------------------------------------------------------------------
__global__ __launch_bounds__(256) void attn_kernel(
        const float* __restrict__ Q, const float* __restrict__ K,
        const float* __restrict__ V, const int* __restrict__ idx,
        float* __restrict__ out) {
    extern __shared__ _Float16 smem[];
    _Float16* Kf  = smem;                  // G*D
    _Float16* Vf  = Kf + G_ * D_;          // G*D
    _Float16* Qf  = Vf + G_ * D_;          // G*D
    _Float16* Pst = Qf + G_ * D_;          // 8 waves * 16*32

    int bid  = blockIdx.x;                 // b*E + e
    int b    = bid >> 4;
    int tid  = threadIdx.x;
    int lane = tid & 31, wave = tid >> 5;
    int halfsel = lane >> 4, ln = lane & 15;

    // ---- gather one expert group's K/V/Q rows into LDS as f16 ----
    {
        int g   = tid;                     // one row per thread (256 rows)
        int src = idx[bid * G_ + g];
        const float* kp = K + ((size_t)b * S_ + src) * D_;
        const float* vp = V + ((size_t)b * S_ + src) * D_;
        const float* qp = Q + ((size_t)b * S_ + src) * D_;
        #pragma unroll 4
        for (int d = 0; d < D_; d += 2) {
            float2 kv = *(const float2*)(kp + d);
            float2 vv = *(const float2*)(vp + d);
            float2 qv = *(const float2*)(qp + d);
            Kf[g * D_ + d] = (_Float16)kv.x; Kf[g * D_ + d + 1] = (_Float16)kv.y;
            Vf[g * D_ + d] = (_Float16)vv.x; Vf[g * D_ + d + 1] = (_Float16)vv.y;
            Qf[g * D_ + d] = (_Float16)qv.x; Qf[g * D_ + d + 1] = (_Float16)qv.y;
        }
    }
    __syncthreads();

    _Float16* Pw = Pst + wave * (16 * 32); // wave-private staging (no barrier needed)

    for (int qi = 0; qi < 2; ++qi) {
        int qt = wave + 8 * qi;            // query tile index 0..15

        v16h aq[4];
        #pragma unroll
        for (int c = 0; c < 4; ++c)
            aq[c] = load_a16x32(Qf, qt * 16 + ln, D_, c * 32, halfsel);

        v8f accs[8];
        #pragma unroll
        for (int dt = 0; dt < 8; ++dt) accs[dt] = (v8f){};
        float m_run[8], l_run[8];
        #pragma unroll
        for (int v = 0; v < 8; ++v) { m_run[v] = -__builtin_inff(); l_run[v] = 0.0f; }

        for (int j = 0; j < 8; ++j) {      // 32-key blocks
            int k0 = j * 32;
            v8f s0 = (v8f){}, s1 = (v8f){};
            #pragma unroll
            for (int c = 0; c < 4; ++c) {
                s0 = wmma16(aq[c], load_bK(Kf, k0,      ln, c, halfsel), s0);
                s1 = wmma16(aq[c], load_bK(Kf, k0 + 16, ln, c, halfsel), s1);
            }

            // online softmax over this 16x32 stripe; C-layout: lane holds one
            // column (ln) of rows M = v + 8*halfsel; a row lives in a 16-lane
            // half-group -> width-16 xor butterflies reduce it.
            float corr[8];
            #pragma unroll
            for (int v = 0; v < 8; ++v) {
                int rg = qt * 16 + halfsel * 8 + v;     // global row in group
                int c0 = k0 + ln, c1 = k0 + 16 + ln;    // global cols
                float x0 = (c0 <= rg) ? s0[v] * SCALE : -__builtin_inff();
                float x1 = (c1 <= rg) ? s1[v] * SCALE : -__builtin_inff();
                float rm = fmaxf(x0, x1);
                #pragma unroll
                for (int off = 8; off >= 1; off >>= 1)
                    rm = fmaxf(rm, __shfl_xor(rm, off, 16));
                float mn = fmaxf(m_run[v], rm);
                float p0 = __expf(x0 - mn);
                float p1 = __expf(x1 - mn);
                float rs = p0 + p1;
                #pragma unroll
                for (int off = 8; off >= 1; off >>= 1)
                    rs += __shfl_xor(rs, off, 16);
                float cr = __expf(m_run[v] - mn);
                l_run[v] = l_run[v] * cr + rs;
                m_run[v] = mn;
                corr[v]  = cr;
                // C-layout -> LDS (row-major 16x32); same-wave DS ops are
                // in-order, so the A-fragment reload below needs no barrier.
                Pw[(halfsel * 8 + v) * 32 + ln]      = (_Float16)p0;
                Pw[(halfsel * 8 + v) * 32 + 16 + ln] = (_Float16)p1;
            }

            #pragma unroll
            for (int dt = 0; dt < 8; ++dt) {
                #pragma unroll
                for (int v = 0; v < 8; ++v) accs[dt][v] *= corr[v];
            }

            v16h pa = load_a16x32(Pw, ln, 32, 0, halfsel);
            #pragma unroll
            for (int dt = 0; dt < 8; ++dt)
                accs[dt] = wmma16(pa, load_bV(Vf, k0, dt * 16, ln, halfsel), accs[dt]);
        }

        // epilogue: divide by row sum, store (B,E,G,D) = (bid, row, d)
        float* ob = out + ((size_t)bid * G_ + qt * 16) * D_;
        #pragma unroll
        for (int v = 0; v < 8; ++v) {
            int rg = halfsel * 8 + v;
            float inv = 1.0f / l_run[v];
            #pragma unroll
            for (int dt = 0; dt < 8; ++dt)
                ob[rg * D_ + dt * 16 + ln] = accs[dt][v] * inv;
        }
    }
}

// ---------------------------------------------------------------------------
extern "C" void kernel_launch(void* const* d_in, const int* in_sizes, int n_in,
                              void* d_out, int out_size, void* d_ws, size_t ws_size,
                              hipStream_t stream) {
    const float* Q     = (const float*)d_in[0];
    const float* K     = (const float*)d_in[1];
    const float* V     = (const float*)d_in[2];
    const float* route = (const float*)d_in[3];
    // d_in[4] = mask(bool G,G): causal triu, recomputed analytically on device.
    float* out = (float*)d_out;

    int* assign = (int*)d_ws;              // B*S ints
    int* idx    = assign + B_ * S_;        // B*E*G ints

    assign_kernel<<<(B_ * S_) / 256, 256, 0, stream>>>(route, assign);
    build_idx_kernel<<<B_ * E_, 32, 0, stream>>>(assign, idx);

    size_t smem = (size_t)(3 * G_ * D_ + 8 * 16 * 32) * 2;   // f16 bytes = 204800
    attn_kernel<<<B_ * E_, 256, smem, stream>>>(Q, K, V, idx, out);
}
